// MixtureOfExperts_81630148428167
// MI455X (gfx1250) — matrix operand order repaired and checked by reading, not compiled, not run on hardware
//
#include <hip/hip_runtime.h>
#include <hip/hip_bf16.h>
#include <math.h>

// Problem constants (match reference)
#define NE 8       // experts
#define DD 1024    // model dim
#define HH 4096    // hidden dim
#define TK 2       // top-k
#define NT 4096    // tokens = B*S
#define TM 64      // tokens per expert block
#define HB 64      // H chunk per iteration
#define XSB 1040   // LDS stride (bf16 elems) for x tile
#define HSB 80     // LDS stride (bf16 elems) for h chunk

typedef float v2f  __attribute__((ext_vector_type(2)));
typedef float v8f  __attribute__((ext_vector_type(8)));
typedef __bf16 v4bf  __attribute__((ext_vector_type(4)));
typedef __bf16 v8bf  __attribute__((ext_vector_type(8)));
typedef __bf16 v16bf __attribute__((ext_vector_type(16)));

// ---------------------------------------------------------------- router ----
__global__ __launch_bounds__(256) void moe_router_k(
    const float* __restrict__ x, const float* __restrict__ rw,
    const float* __restrict__ rb, float* __restrict__ logits_out,
    int* __restrict__ topk_idx, float* __restrict__ topk_gate,
    int* __restrict__ counts) {
  const int wave = threadIdx.x >> 5;
  const int lane = threadIdx.x & 31;
  const int t = blockIdx.x * 8 + wave;          // grid = NT/8 blocks
  const float* xrow = x + (size_t)t * DD;

  float acc[NE];
#pragma unroll
  for (int e = 0; e < NE; ++e) acc[e] = 0.f;

  for (int d = lane; d < DD; d += 32) {
    float xv = xrow[d];
    const float4* r4 = (const float4*)(rw + (size_t)d * NE);
    float4 a = r4[0], b = r4[1];
    acc[0] += xv * a.x; acc[1] += xv * a.y; acc[2] += xv * a.z; acc[3] += xv * a.w;
    acc[4] += xv * b.x; acc[5] += xv * b.y; acc[6] += xv * b.z; acc[7] += xv * b.w;
  }
#pragma unroll
  for (int off = 16; off > 0; off >>= 1) {
#pragma unroll
    for (int e = 0; e < NE; ++e) acc[e] += __shfl_xor(acc[e], off, 32);
  }

  if (lane == 0) {
    float lg[NE];
#pragma unroll
    for (int e = 0; e < NE; ++e) {
      lg[e] = acc[e] + rb[e];
      logits_out[(size_t)t * NE + e] = lg[e];
    }
    int i0 = 0;
#pragma unroll
    for (int e = 1; e < NE; ++e) if (lg[e] > lg[i0]) i0 = e;
    int i1 = (i0 == 0) ? 1 : 0;
#pragma unroll
    for (int e = 0; e < NE; ++e) if (e != i0 && lg[e] > lg[i1]) i1 = e;
    float e1 = __expf(lg[i1] - lg[i0]);          // <= 1, safe
    float inv = 1.f / (1.f + e1);
    topk_idx[t * TK + 0] = i0;  topk_gate[t * TK + 0] = inv;
    topk_idx[t * TK + 1] = i1;  topk_gate[t * TK + 1] = e1 * inv;
    atomicAdd(&counts[i0], 1);
    atomicAdd(&counts[i1], 1);
  }
}

// ------------------------------------------------------------------ scan ----
__global__ void moe_scan_k(const int* __restrict__ counts, int* __restrict__ offsets) {
  if (threadIdx.x == 0 && blockIdx.x == 0) {
    int s = 0;
#pragma unroll
    for (int e = 0; e < NE; ++e) { offsets[e] = s; s += counts[e]; }
    offsets[NE] = s;
  }
}

// --------------------------------------------------------------- scatter ----
__global__ __launch_bounds__(256) void moe_scatter_k(
    const int* __restrict__ topk_idx, const int* __restrict__ offsets,
    int* __restrict__ cursor, int* __restrict__ slot_tok, int* __restrict__ slot_of) {
  int t = blockIdx.x * 256 + threadIdx.x;
  if (t >= NT) return;
#pragma unroll
  for (int k = 0; k < TK; ++k) {
    int e = topk_idx[t * TK + k];
    int pos = atomicAdd(&cursor[e], 1);
    int slot = offsets[e] + pos;
    slot_tok[slot] = t;
    slot_of[t * TK + k] = slot;
  }
}

// ------------------------------------------------ x fp32 -> bf16 rows ------
__global__ __launch_bounds__(256) void moe_cvt_x_k(
    const float* __restrict__ x, __bf16* __restrict__ xb) {
  int gid = blockIdx.x * 256 + threadIdx.x;          // NT*DD/4 groups
  float4 v = ((const float4*)x)[gid];
  v4bf o = { (__bf16)v.x, (__bf16)v.y, (__bf16)v.z, (__bf16)v.w };
  ((v4bf*)xb)[gid] = o;
}

// -------------------- weights fp32 -> bf16 pre-swizzled into B-frag order --
// For each 32x16 (KxN) tile: lane l owns a contiguous 32B blob =
// W[kt*32 + 8*(l>>4) + j + (j>=8 ? 8 : 0)][nt*16 + (l&15)], j = 0..15  (bf16 B-matrix layout)
__global__ __launch_bounds__(256) void moe_swz_w_k(
    const float* __restrict__ w, __bf16* __restrict__ wout,
    int Kdim, int Ndim) {
  int gid = blockIdx.x * 256 + threadIdx.x;          // blob id
  int l    = gid & 31;
  int tile = gid >> 5;
  int ntiles = Ndim >> 4;
  int ktiles = Kdim >> 5;
  int nt = tile % ntiles;
  int kt = (tile / ntiles) % ktiles;
  int e  = tile / (ntiles * ktiles);
  const float* we = w + (size_t)e * Kdim * Ndim;
  int n  = nt * 16 + (l & 15);
  int k0 = kt * 32 + (l >> 4) * 8;
  __bf16 blob[16];
#pragma unroll
  for (int j = 0; j < 16; ++j) blob[j] = (__bf16)we[(size_t)(k0 + j + (j >= 8 ? 8 : 0)) * Ndim + n];
  ((v16bf*)wout)[gid] = *(v16bf*)blob;
}

// ----------------------------------------------------- fused expert FFN -----
// Block: 512 threads (16 waves), 64-token tile of one expert. bf16 WMMA.
__global__ __launch_bounds__(512) void moe_expert_k(
    const __bf16* __restrict__ w1s, const float* __restrict__ b1,
    const __bf16* __restrict__ w2s, const float* __restrict__ b2,
    const float* __restrict__ gamma, const float* __restrict__ beta,
    const float* __restrict__ x, const __bf16* __restrict__ xb,
    const int* __restrict__ counts, const int* __restrict__ offsets,
    const int* __restrict__ slot_tok, float* __restrict__ eo) {
  const int e = blockIdx.y;
  const int cnt = counts[e];
  const int r0 = blockIdx.x * TM;
  if (r0 >= cnt) return;
  const int base = offsets[e];

  extern __shared__ __bf16 smem[];
  __bf16* xsb = smem;                        // TM * XSB bf16
  __bf16* hsb = smem + TM * XSB;             // TM * HSB bf16
  float*  ps  = (float*)(hsb + TM * HSB);    // TM * 4
  float*  ps2 = ps  + TM * 4;                // TM * 4
  float*  mu  = ps2 + TM * 4;                // TM
  float*  rsd = mu  + TM;                    // TM
  int*    ts  = (int*)(rsd + TM);            // TM

  const int tid  = threadIdx.x;
  const int wv   = tid >> 5;                 // 0..15
  const int lane = tid & 31;
  const int half = lane >> 4;                // 0/1
  const int ln   = lane & 15;
  const int rq   = wv & 3;                   // row quad (16 rows each)
  const int cg   = wv >> 2;                  // col group

  if (tid < TM) ts[tid] = (r0 + tid < cnt) ? slot_tok[base + r0 + tid] : -1;
  __syncthreads();

  // ---- stage X tile (bf16) via async global->LDS copies (16B granules)
  {
    uint32_t lds0 = (uint32_t)(uintptr_t)xsb;
    for (int i = tid; i < TM * (DD / 8); i += 512) {   // 64 rows * 128 chunks
      int row = i >> 7;
      int c16 = i & 127;
      int tk  = ts[row];
      if (tk >= 0) {
        uint32_t loff = lds0 + (uint32_t)(row * XSB + c16 * 8) * 2;
        const __bf16* g = xb + (size_t)tk * DD + c16 * 8;
        asm volatile("global_load_async_to_lds_b128 %0, %1, off"
                     :: "v"(loff), "v"(g) : "memory");
      } else {
        *(float4*)(xsb + row * XSB + c16 * 8) = make_float4(0.f, 0.f, 0.f, 0.f);
      }
    }
    asm volatile("s_wait_asynccnt 0x0" ::: "memory");
  }
  __syncthreads();

  const __bf16* w1e = w1s + (size_t)e * DD * HH;
  const __bf16* w2e = w2s + (size_t)e * HH * DD;

  v8f yacc[16];
#pragma unroll
  for (int j = 0; j < 16; ++j) yacc[j] = (v8f){0.f,0.f,0.f,0.f,0.f,0.f,0.f,0.f};

  for (int c = 0; c < HH / HB; ++c) {
    // ---- GEMM1: wave computes h tile rows [rq*16,+16), cols [c*64+cg*16,+16)
    const int nt1 = c * (HB / 16) + cg;                // n-tile index in w1s
    v8f hacc = (v8f){0.f,0.f,0.f,0.f,0.f,0.f,0.f,0.f};
    for (int kb = 0; kb < DD; kb += 32) {
      const __bf16* arow = xsb + (rq * 16 + ln) * XSB + kb + 8 * half;
      v8bf a0 = *(const v8bf*)(arow);
      v8bf a1 = *(const v8bf*)(arow + 16);
      v16bf a = __builtin_shufflevector(a0, a1, 0,1,2,3,4,5,6,7,8,9,10,11,12,13,14,15);
      const int kt = kb >> 5;
      v16bf b = *(const v16bf*)(w1e + ((size_t)(kt * (HH >> 4) + nt1) << 9) + (lane << 4));
      hacc = __builtin_amdgcn_wmma_f32_16x16x32_bf16(
          false, a, false, b, (short)0, hacc, false, false);
    }
    // bias + exact GELU -> bf16 into LDS h chunk
#pragma unroll
    for (int i = 0; i < 8; ++i) {
      int mrow = rq * 16 + (half ? (i + 8) : i);
      int hcol = c * HB + cg * 16 + ln;
      float hv = hacc[i] + b1[(size_t)e * HH + hcol];
      float g  = 0.5f * hv * (1.0f + erff(hv * 0.70710678f));
      hsb[mrow * HSB + cg * 16 + ln] = (__bf16)g;
    }
    __syncthreads();

    // prefetch next chunk's w1 blobs
    if (c + 1 < HH / HB)
      __builtin_prefetch(w1e + ((size_t)((c + 1) * (HB / 16) + cg) << 9) + (lane << 4), 0, 0);

    // ---- GEMM2: wave owns rows [rq*16,+16), d-cols [cg*256,+256) (16 tiles)
    for (int kb2 = 0; kb2 < HB; kb2 += 32) {
      const __bf16* arow = hsb + (rq * 16 + ln) * HSB + kb2 + 8 * half;
      v8bf a0 = *(const v8bf*)(arow);
      v8bf a1 = *(const v8bf*)(arow + 16);
      v16bf a = __builtin_shufflevector(a0, a1, 0,1,2,3,4,5,6,7,8,9,10,11,12,13,14,15);
      const int kt2 = c * (HB / 32) + (kb2 >> 5);
#pragma unroll
      for (int j = 0; j < 16; ++j) {
        const int nt2 = cg * 16 + j;
        v16bf b = *(const v16bf*)(w2e + ((size_t)(kt2 * (DD >> 4) + nt2) << 9) + (lane << 4));
        yacc[j] = __builtin_amdgcn_wmma_f32_16x16x32_bf16(
            false, a, false, b, (short)0, yacc[j], false, false);
      }
    }
    __syncthreads();   // hsb reused next chunk
  }

  // ---- epilogue pass 1: residual (fp32 x) + bias, per-row partial sums
  float s1v[8], s2v[8];
#pragma unroll
  for (int i = 0; i < 8; ++i) { s1v[i] = 0.f; s2v[i] = 0.f; }
#pragma unroll
  for (int j = 0; j < 16; ++j) {
    int dcol = cg * 256 + j * 16 + ln;
    float bb = b2[(size_t)e * DD + dcol];
#pragma unroll
    for (int i = 0; i < 8; ++i) {
      int mrow = rq * 16 + (half ? (i + 8) : i);
      int tk = ts[mrow];
      float xv = (tk >= 0) ? x[(size_t)tk * DD + dcol] : 0.f;
      float r = (tk >= 0) ? (yacc[j][i] + bb + xv) : 0.f;
      yacc[j][i] = r;
      s1v[i] += r;  s2v[i] += r * r;
    }
  }
#pragma unroll
  for (int i = 0; i < 8; ++i) {
    float s = s1v[i], s2 = s2v[i];
#pragma unroll
    for (int off = 8; off > 0; off >>= 1) {     // reduce 16 lanes within half
      s  += __shfl_xor(s,  off, 32);
      s2 += __shfl_xor(s2, off, 32);
    }
    if (ln == 0) {
      int mrow = rq * 16 + (half ? (i + 8) : i);
      ps [mrow * 4 + cg] = s;
      ps2[mrow * 4 + cg] = s2;
    }
  }
  __syncthreads();
  if (tid < TM) {
    float s  = ps [tid*4] + ps [tid*4+1] + ps [tid*4+2] + ps [tid*4+3];
    float s2 = ps2[tid*4] + ps2[tid*4+1] + ps2[tid*4+2] + ps2[tid*4+3];
    float mean = s * (1.f / DD);
    mu[tid]  = mean;
    rsd[tid] = rsqrtf(s2 * (1.f / DD) - mean * mean + 1e-5f);
  }
  __syncthreads();

  // ---- epilogue pass 2: normalize + affine, store per-slot expert output
#pragma unroll
  for (int j = 0; j < 16; ++j) {
    int dcol = cg * 256 + j * 16 + ln;
    float gm = gamma[(size_t)e * DD + dcol];
    float bt = beta [(size_t)e * DD + dcol];
#pragma unroll
    for (int i = 0; i < 8; ++i) {
      int mrow = rq * 16 + (half ? (i + 8) : i);
      if (ts[mrow] >= 0) {
        float v = (yacc[j][i] - mu[mrow]) * rsd[mrow] * gm + bt;
        eo[(size_t)(base + r0 + mrow) * DD + dcol] = v;
      }
    }
  }
}

// --------------------------------------------------------------- combine ----
__global__ __launch_bounds__(256) void moe_combine_k(
    const float* __restrict__ eo, const int* __restrict__ slot_of,
    const float* __restrict__ topk_gate, float* __restrict__ out) {
  int gid = blockIdx.x * 256 + threadIdx.x;      // grid covers NT*DD
  int t  = gid >> 10;
  int d0 = gid & (DD - 1);
  int s0 = slot_of[t * TK + 0], s1 = slot_of[t * TK + 1];
  float g0 = topk_gate[t * TK + 0], g1 = topk_gate[t * TK + 1];
  out[gid] = g0 * eo[(size_t)s0 * DD + d0] + g1 * eo[(size_t)s1 * DD + d0];
}

// ---------------------------------------------------------------- launch ----
extern "C" void kernel_launch(void* const* d_in, const int* in_sizes, int n_in,
                              void* d_out, int out_size, void* d_ws, size_t ws_size,
                              hipStream_t stream) {
  const float* x        = (const float*)d_in[0];
  const float* router_w = (const float*)d_in[1];
  const float* router_b = (const float*)d_in[2];
  const float* w1       = (const float*)d_in[3];
  const float* b1       = (const float*)d_in[4];
  const float* w2       = (const float*)d_in[5];
  const float* b2       = (const float*)d_in[6];
  const float* gamma    = (const float*)d_in[7];
  const float* beta     = (const float*)d_in[8];

  float* out    = (float*)d_out;                       // [NT, DD]
  float* logits = out + (size_t)NT * DD;               // [NT, NE]

  char* ws = (char*)d_ws;
  int*    topk_idx  = (int*)(ws + 0);                  // NT*TK ints
  float*  topk_gate = (float*)(ws + 32768);            // NT*TK floats
  int*    slot_tok  = (int*)(ws + 65536);              // NT*TK ints
  int*    slot_of   = (int*)(ws + 98304);              // NT*TK ints
  int*    counts    = (int*)(ws + 131072);             // NE ints
  int*    cursor    = (int*)(ws + 131104);             // NE ints
  int*    offsets   = (int*)(ws + 131136);             // NE+1 ints
  __bf16* xb        = (__bf16*)(ws + 131584);          // NT*DD bf16      (8 MB)
  __bf16* w1s       = (__bf16*)(ws + 131584 + 8388608);            // 64 MB
  __bf16* w2s       = (__bf16*)(ws + 131584 + 8388608 + 67108864); // 64 MB
  float*  eo        = (float*)(ws + 131584 + 8388608 + 2*67108864); // 33.5 MB

  hipMemsetAsync(counts, 0, 64, stream);   // counts + cursor

  moe_cvt_x_k<<<(NT * DD / 4) / 256, 256, 0, stream>>>(x, xb);
  moe_swz_w_k<<<(NE * (DD/32) * (HH/16) * 32) / 256, 256, 0, stream>>>(w1, w1s, DD, HH);
  moe_swz_w_k<<<(NE * (HH/32) * (DD/16) * 32) / 256, 256, 0, stream>>>(w2, w2s, HH, DD);

  moe_router_k<<<NT / 8, 256, 0, stream>>>(x, router_w, router_b, logits,
                                           topk_idx, topk_gate, counts);
  moe_scan_k<<<1, 1, 0, stream>>>(counts, offsets);
  moe_scatter_k<<<NT / 256, 256, 0, stream>>>(topk_idx, offsets, cursor,
                                              slot_tok, slot_of);

  size_t lds_bytes = (size_t)(TM * XSB + TM * HSB) * 2   // bf16 tiles
                   + (size_t)(TM * 4 + TM * 4 + TM + TM) * 4
                   + (size_t)TM * 4;
  moe_expert_k<<<dim3(NT / TM, NE), 512, lds_bytes, stream>>>(
      w1s, b1, w2s, b2, gamma, beta, x, xb, counts, offsets, slot_tok, eo);

  moe_combine_k<<<(NT * DD) / 256, 256, 0, stream>>>(eo, slot_of, topk_gate, out);
}
